// Dynamic_Cons_17540646437049
// MI455X (gfx1250) — compile-verified, matrix operand order checked by reading, not run on hardware
//
#include <hip/hip_runtime.h>
#include <hip/hip_bf16.h>
#include <cstddef>
#include <cstdint>

// ---------------------------------------------------------------------------
// Problem constants (from reference)
// ---------------------------------------------------------------------------
#define N_TOK   16384
#define DIM_    1024
#define HEAD_   16
#define K_G_    256
#define TOPK_   16
#define HDIM_   64

typedef __attribute__((ext_vector_type(16))) _Float16 v16h;
typedef __attribute__((ext_vector_type(8)))  float    v8f;
typedef __attribute__((ext_vector_type(4)))  unsigned u32x4;
typedef __attribute__((ext_vector_type(8)))  int      i32x8;
typedef __attribute__((ext_vector_type(4)))  int      i32x4;

union FragU { v16h h; unsigned u[8]; };

__device__ __forceinline__ float sigmoidf_(float x) {
    return 1.0f / (1.0f + __expf(-x));
}

// ---------------------------------------------------------------------------
// TDM: issue a 2D tensor_load_to_lds of a (128 rows x 32 halves) tile.
//   - data_size = 2 bytes, tile_dim0 = 32 elems (64B rows), tile_dim1 = 128
//   - LDS pad: after every 16 dwords (one row) insert 4 dwords -> 40-half rows
//   - D# group0: count=1, lds_addr, global_addr, type=2
//   - D# group1: data_size/pad fields, tensor dims, tile dims, dim0 stride
// ---------------------------------------------------------------------------
__device__ __forceinline__ void tdm_load_tile_2d(unsigned lds_off,
                                                 const _Float16* gptr,
                                                 unsigned rows_total,
                                                 unsigned stride_elems)
{
    const unsigned long long ga = (unsigned long long)(uintptr_t)gptr;
    u32x4 g0;
    g0[0] = 1u;                                        // count=1, user mode
    g0[1] = lds_off;                                   // lds_addr (bytes)
    g0[2] = (unsigned)(ga & 0xFFFFFFFFull);            // global_addr[31:0]
    g0[3] = (unsigned)((ga >> 32) & 0x01FFFFFFull)     // global_addr[56:32]
          | (2u << 30);                                // type = 2 (image)

    i32x8 g1;
    // data_size=1(2B)<<16 | pad_enable<<20 | pad_interval=3(16 dw)<<22 | pad_amount=3(4 dw)<<25
    g1[0] = (int)(0x10000u | (1u << 20) | (3u << 22) | (3u << 25));
    g1[1] = (int)((stride_elems & 0xFFFFu) << 16);     // tensor_dim0[15:0] @ bit48
    g1[2] = (int)((stride_elems >> 16) | ((rows_total & 0xFFFFu) << 16));
    g1[3] = (int)(((rows_total >> 16) & 0xFFFFu) | (32u << 16));  // tile_dim0 = 32
    g1[4] = 128;                                       // tile_dim1=128, tile_dim2=0
    g1[5] = (int)stride_elems;                         // tensor_dim0_stride[31:0]
    g1[6] = 0;                                         // stride[47:32]=0, dim1_stride lo=0
    g1[7] = 0;

    const i32x4 z4 = {0, 0, 0, 0};
#if __has_include(<hip/amd_detail/amd_gfx1250_TDM.h>)
    const i32x8 z8 = {0, 0, 0, 0, 0, 0, 0, 0};
    __builtin_amdgcn_tensor_load_to_lds(g0, g1, z4, z4, z8, 0);
#else
    __builtin_amdgcn_tensor_load_to_lds(g0, g1, z4, z4, 0);
#endif
}

// ---------------------------------------------------------------------------
// f32 -> f16 elementwise convert (8 elems/thread, vectorized ld/st)
// ---------------------------------------------------------------------------
__global__ __launch_bounds__(256)
void cvt_f16_kernel(const float* __restrict__ in, _Float16* __restrict__ out)
{
    const size_t i = ((size_t)blockIdx.x * 256 + threadIdx.x) * 8;
    const float4 a = *(const float4*)(in + i);
    const float4 b = *(const float4*)(in + i + 4);
    union { _Float16 h[8]; uint4 u; } t;
    t.h[0] = (_Float16)a.x; t.h[1] = (_Float16)a.y;
    t.h[2] = (_Float16)a.z; t.h[3] = (_Float16)a.w;
    t.h[4] = (_Float16)b.x; t.h[5] = (_Float16)b.y;
    t.h[6] = (_Float16)b.z; t.h[7] = (_Float16)b.w;
    *(uint4*)(out + i) = t.u;
}

// ---------------------------------------------------------------------------
// f32 [R][C] -> f16 [C][R] transpose-convert (for W matrices)
// ---------------------------------------------------------------------------
__global__ __launch_bounds__(256)
void cvtT_f16_kernel(const float* __restrict__ in, _Float16* __restrict__ out,
                     int R, int C)
{
    __shared__ float t[32][33];
    const int bx = blockIdx.x * 32;  // over R
    const int by = blockIdx.y * 32;  // over C
    const int tx = threadIdx.x & 31;
    const int ty = threadIdx.x >> 5; // 0..7
#pragma unroll
    for (int i = 0; i < 32; i += 8)
        t[ty + i][tx] = in[(size_t)(bx + ty + i) * C + by + tx];
    __syncthreads();
#pragma unroll
    for (int i = 0; i < 32; i += 8)
        out[(size_t)(by + ty + i) * R + bx + tx] = (_Float16)t[tx][ty + i];
}

// ---------------------------------------------------------------------------
// WMMA GEMM with TDM-staged, double-buffered LDS tiles.
// C[M x Nc] = A16[M x K] * B16t[Nc x K]^T + bias
// OUT_HALF: write f16 (for q), else f32.
// Block tile 128x128, K-step 32, 8 waves; wave w owns rows 16w..16w+15.
// ---------------------------------------------------------------------------
template <bool OUT_HALF>
__global__ __launch_bounds__(256)
void gemm_wmma_tdm(const _Float16* __restrict__ A16,
                   const _Float16* __restrict__ B16t,
                   const float* __restrict__ bias,
                   void* __restrict__ Cout,
                   int M, int K, int Nc)
{
    __shared__ _Float16 As[2][128 * 40];   // 40-half rows (TDM pad)
    __shared__ _Float16 Bs[2][128 * 40];

    const int tid  = threadIdx.x;
    const int wave = tid >> 5;
    const int lane = tid & 31;
    const int half = lane >> 4;
    const int l16  = lane & 15;
    const int rowBase = blockIdx.x * 128;
    const int colBase = blockIdx.y * 128;
    const int steps = K / 32;

    const unsigned ldsA0 = (unsigned)(uintptr_t)&As[0][0];
    const unsigned ldsA1 = (unsigned)(uintptr_t)&As[1][0];
    const unsigned ldsB0 = (unsigned)(uintptr_t)&Bs[0][0];
    const unsigned ldsB1 = (unsigned)(uintptr_t)&Bs[1][0];

    const v8f vzero = {0.f,0.f,0.f,0.f,0.f,0.f,0.f,0.f};
    v8f acc[8];
#pragma unroll
    for (int c = 0; c < 8; ++c) acc[c] = vzero;

    // prologue: stage tile 0 into buffer 0
    if (wave == 0) {
        tdm_load_tile_2d(ldsA0, A16 + (size_t)rowBase * K, (unsigned)M, (unsigned)K);
        tdm_load_tile_2d(ldsB0, B16t + (size_t)colBase * K, (unsigned)Nc, (unsigned)K);
    }

    for (int i = 0; i < steps; ++i) {
        const int buf = i & 1;
        // prefetch next tile pair into the other buffer, then drain to <=2
        if (wave == 0) {
            if (i + 1 < steps) {
                const int k1 = (i + 1) * 32;
                const unsigned la = (buf ? ldsA0 : ldsA1);
                const unsigned lb = (buf ? ldsB0 : ldsB1);
                tdm_load_tile_2d(la, A16 + (size_t)rowBase * K + k1, (unsigned)M, (unsigned)K);
                tdm_load_tile_2d(lb, B16t + (size_t)colBase * K + k1, (unsigned)Nc, (unsigned)K);
                __builtin_amdgcn_s_wait_tensorcnt((short)2);
            } else {
                __builtin_amdgcn_s_wait_tensorcnt((short)0);
            }
        }
        __syncthreads();

        // A fragment: lane holds row m = 16*wave + l16 (ISA 16-bit A layout)
        FragU afr;
        {
            const unsigned* arow = (const unsigned*)&As[buf][(wave * 16 + l16) * 40];
#pragma unroll
            for (int p = 0; p < 8; ++p) {
                const int dw = 4 * half + p + ((p >= 4) ? 4 : 0);
                afr.u[p] = arow[dw];
            }
        }
#pragma unroll
        for (int c = 0; c < 8; ++c) {
            FragU bfr;
            const unsigned* brow = (const unsigned*)&Bs[buf][(c * 16 + l16) * 40];
#pragma unroll
            for (int p = 0; p < 8; ++p) {
                const int dw = 4 * half + p + ((p >= 4) ? 4 : 0);
                bfr.u[p] = brow[dw];
            }
            acc[c] = __builtin_amdgcn_wmma_f32_16x16x32_f16(
                false, afr.h, false, bfr.h, (short)0, acc[c], false, false);
        }
        __syncthreads();
    }

    // epilogue: D layout VGPR i -> (M = i + 8*half, N = l16)
#pragma unroll
    for (int c = 0; c < 8; ++c) {
        const int n = colBase + c * 16 + l16;
        const float bvv = bias ? bias[n] : 0.0f;
#pragma unroll
        for (int i = 0; i < 8; ++i) {
            const int m = rowBase + wave * 16 + i + 8 * half;
            const float v = acc[c][i] + bvv;
            if (OUT_HALF)
                ((_Float16*)Cout)[(size_t)m * Nc + n] = (_Float16)v;
            else
                ((float*)Cout)[(size_t)m * Nc + n] = v;
        }
    }
}

// ---------------------------------------------------------------------------
// attn[n][h][r] = (1/8) * sum_d qh[n, h*64+d] * kph[r, h, d]
// Per-head WMMA GEMM with TDM staging (K=64 -> 2 steps).
// ---------------------------------------------------------------------------
__global__ __launch_bounds__(256)
void attn_wmma_tdm(const _Float16* __restrict__ qh, const _Float16* __restrict__ kph,
                   float* __restrict__ attn)
{
    __shared__ _Float16 As[2][128 * 40];
    __shared__ _Float16 Bs[2][128 * 40];

    const int tid  = threadIdx.x;
    const int wave = tid >> 5;
    const int lane = tid & 31;
    const int half = lane >> 4;
    const int l16  = lane & 15;
    const int rowBase = blockIdx.x * 128;   // over n
    const int colBase = blockIdx.y * 128;   // over r
    const int h = blockIdx.z;

    const unsigned ldsA0 = (unsigned)(uintptr_t)&As[0][0];
    const unsigned ldsA1 = (unsigned)(uintptr_t)&As[1][0];
    const unsigned ldsB0 = (unsigned)(uintptr_t)&Bs[0][0];
    const unsigned ldsB1 = (unsigned)(uintptr_t)&Bs[1][0];

    const _Float16* aBase = qh + (size_t)rowBase * DIM_ + h * HDIM_;
    const _Float16* bBase = kph + (size_t)colBase * (HEAD_ * HDIM_) + h * HDIM_;

    const v8f vzero = {0.f,0.f,0.f,0.f,0.f,0.f,0.f,0.f};
    v8f acc[8];
#pragma unroll
    for (int c = 0; c < 8; ++c) acc[c] = vzero;

    if (wave == 0) {
        tdm_load_tile_2d(ldsA0, aBase, (unsigned)N_TOK, (unsigned)DIM_);
        tdm_load_tile_2d(ldsB0, bBase, (unsigned)K_G_, (unsigned)(HEAD_ * HDIM_));
    }

#pragma unroll
    for (int i = 0; i < 2; ++i) {
        const int buf = i & 1;
        if (wave == 0) {
            if (i == 0) {
                tdm_load_tile_2d(ldsA1, aBase + 32, (unsigned)N_TOK, (unsigned)DIM_);
                tdm_load_tile_2d(ldsB1, bBase + 32, (unsigned)K_G_, (unsigned)(HEAD_ * HDIM_));
                __builtin_amdgcn_s_wait_tensorcnt((short)2);
            } else {
                __builtin_amdgcn_s_wait_tensorcnt((short)0);
            }
        }
        __syncthreads();

        FragU afr;
        {
            const unsigned* arow = (const unsigned*)&As[buf][(wave * 16 + l16) * 40];
#pragma unroll
            for (int p = 0; p < 8; ++p) {
                const int dw = 4 * half + p + ((p >= 4) ? 4 : 0);
                afr.u[p] = arow[dw];
            }
        }
#pragma unroll
        for (int c = 0; c < 8; ++c) {
            FragU bfr;
            const unsigned* brow = (const unsigned*)&Bs[buf][(c * 16 + l16) * 40];
#pragma unroll
            for (int p = 0; p < 8; ++p) {
                const int dw = 4 * half + p + ((p >= 4) ? 4 : 0);
                bfr.u[p] = brow[dw];
            }
            acc[c] = __builtin_amdgcn_wmma_f32_16x16x32_f16(
                false, afr.h, false, bfr.h, (short)0, acc[c], false, false);
        }
        __syncthreads();
    }

#pragma unroll
    for (int c = 0; c < 8; ++c) {
        const int rr = colBase + c * 16 + l16;
#pragma unroll
        for (int i = 0; i < 8; ++i) {
            const int m = rowBase + wave * 16 + i + 8 * half;
            attn[((size_t)m * HEAD_ + h) * K_G_ + rr] = acc[c][i] * 0.125f;
        }
    }
}

// ---------------------------------------------------------------------------
// tiled transpose attn [N][H*K_G] -> attn_t [H*K_G][N]
// ---------------------------------------------------------------------------
__global__ __launch_bounds__(256)
void transpose_kernel(const float* __restrict__ in, float* __restrict__ out)
{
    __shared__ float t[32][33];
    const int bx = blockIdx.x * 32;  // n base
    const int by = blockIdx.y * 32;  // hr base
    const int tx = threadIdx.x & 31;
    const int ty = threadIdx.x >> 5;
#pragma unroll
    for (int i = 0; i < 32; i += 8)
        t[ty + i][tx] = in[(size_t)(bx + ty + i) * (HEAD_ * K_G_) + by + tx];
    __syncthreads();
#pragma unroll
    for (int i = 0; i < 32; i += 8)
        out[(size_t)(by + ty + i) * N_TOK + bx + tx] = t[tx][ty + i];
}

// ---------------------------------------------------------------------------
// per (n,h) row: top-16 over keys by sigmoid(attn)+bias; one wave per row.
// ---------------------------------------------------------------------------
__global__ __launch_bounds__(256)
void topk_rows_kernel(const float* __restrict__ attn, const float* __restrict__ bias,
                      float* __restrict__ sel_w, int* __restrict__ sel_i,
                      float* __restrict__ out_idx)
{
    const int wave = threadIdx.x >> 5;
    const int lane = threadIdx.x & 31;
    const size_t row = (size_t)blockIdx.x * 8 + wave;   // over N*HEAD
    const float* arow = attn + row * K_G_;
    const float* brow = bias + row * K_G_;

    float a[8], s[8];
#pragma unroll
    for (int j = 0; j < 8; ++j) {
        const float av = arow[lane + 32 * j];
        a[j] = av;
        s[j] = sigmoidf_(av) + brow[lane + 32 * j];
    }

    float sel_a[16]; int sel_r[16];
#pragma unroll
    for (int t = 0; t < 16; ++t) {
        float bv = -1e30f; int bi = 1 << 30; float ba = 0.f;
#pragma unroll
        for (int j = 0; j < 8; ++j) {
            const int idx = lane + 32 * j;
            if (s[j] > bv || (s[j] == bv && idx < bi)) { bv = s[j]; bi = idx; ba = a[j]; }
        }
#pragma unroll
        for (int off = 16; off >= 1; off >>= 1) {
            const float ov = __shfl_xor(bv, off, 32);
            const int   oi = __shfl_xor(bi, off, 32);
            const float oa = __shfl_xor(ba, off, 32);
            if (ov > bv || (ov == bv && oi < bi)) { bv = ov; bi = oi; ba = oa; }
        }
        sel_a[t] = ba; sel_r[t] = bi;
#pragma unroll
        for (int j = 0; j < 8; ++j)
            if (lane + 32 * j == bi) s[j] = -1e30f;
    }

    float mx = -1e30f;
#pragma unroll
    for (int t = 0; t < 16; ++t) mx = fmaxf(mx, sel_a[t]);
    float w[16]; float sum = 0.f;
#pragma unroll
    for (int t = 0; t < 16; ++t) { w[t] = __expf(sel_a[t] - mx); sum += w[t]; }
    const float inv = 1.f / sum;

    if (lane == 0) {
#pragma unroll
        for (int t = 0; t < 16; ++t) {
            sel_w[row * TOPK_ + t]   = w[t] * inv;
            sel_i[row * TOPK_ + t]   = sel_r[t];
            out_idx[row * TOPK_ + t] = (float)sel_r[t];
        }
    }
}

// ---------------------------------------------------------------------------
// per (h,r): top-16 over queries; v_r[r][h][d] = sum sigmoid(a+emb)*xv
// ---------------------------------------------------------------------------
__global__ __launch_bounds__(256)
void topk_cols_vr_kernel(const float* __restrict__ attn_t, const float* __restrict__ xv,
                         const float* __restrict__ emb, float* __restrict__ v_r)
{
    __shared__ float rv[256];
    __shared__ int   ri[256];
    __shared__ float sel_a[16];
    __shared__ int   sel_n[16];

    const int h = blockIdx.x >> 8;
    const int r = blockIdx.x & 255;
    const int tid = threadIdx.x;
    const float* col = attn_t + (size_t)(h * K_G_ + r) * N_TOK;

    float v64[64];
#pragma unroll
    for (int j = 0; j < 64; ++j) v64[j] = col[tid + 256 * j];

    float pv = 1e30f; int pn = -1;
    for (int t = 0; t < 16; ++t) {
        float bv = -1e30f; int bn = 1 << 30;
#pragma unroll
        for (int j = 0; j < 64; ++j) {
            const int n = tid + 256 * j;
            const bool elig = (v64[j] < pv) || (v64[j] == pv && n > pn);
            if (elig && (v64[j] > bv || (v64[j] == bv && n < bn))) { bv = v64[j]; bn = n; }
        }
        rv[tid] = bv; ri[tid] = bn;
        __syncthreads();
        for (int s = 128; s > 0; s >>= 1) {
            if (tid < s) {
                if (rv[tid + s] > rv[tid] ||
                    (rv[tid + s] == rv[tid] && ri[tid + s] < ri[tid])) {
                    rv[tid] = rv[tid + s]; ri[tid] = ri[tid + s];
                }
            }
            __syncthreads();
        }
        pv = rv[0]; pn = ri[0];
        if (tid == 0) { sel_a[t] = pv; sel_n[t] = pn; }
        __syncthreads();
    }

    const float e = emb[h * K_G_ + r];
    if (tid < HDIM_) {
        float acc = 0.f;
        for (int t = 0; t < 16; ++t) {
            const float g = sigmoidf_(sel_a[t] + e);
            acc += g * xv[(size_t)sel_n[t] * DIM_ + h * HDIM_ + tid];
        }
        v_r[((size_t)r * HEAD_ + h) * HDIM_ + tid] = acc;
    }
}

// ---------------------------------------------------------------------------
// v[n] = sig(alpha[h])*xv + sig(beta[h]) * sum_j sel_w * v_r[sel_i]
// ---------------------------------------------------------------------------
__global__ __launch_bounds__(256)
void final_kernel(const float* __restrict__ xv, const float* __restrict__ v_r,
                  const float* __restrict__ sel_w, const int* __restrict__ sel_i,
                  const float* __restrict__ alpha, const float* __restrict__ beta,
                  float* __restrict__ out)
{
    __shared__ float lw[16][16];
    __shared__ int   lr[16][16];
    const int n = blockIdx.x;
    const int tid = threadIdx.x;
    {
        const int hh = tid >> 4, jj = tid & 15;
        lw[hh][jj] = sel_w[((size_t)n * HEAD_ + hh) * TOPK_ + jj];
        lr[hh][jj] = sel_i[((size_t)n * HEAD_ + hh) * TOPK_ + jj];
    }
    __syncthreads();
#pragma unroll
    for (int c = 0; c < 4; ++c) {
        const int idx = tid + 256 * c;
        const int h = idx >> 6, d = idx & 63;
        const float sa = sigmoidf_(alpha[h]);
        const float sb = sigmoidf_(beta[h]);
        float acc = 0.f;
#pragma unroll
        for (int j = 0; j < 16; ++j)
            acc += lw[h][j] * v_r[((size_t)lr[h][j] * HEAD_ + h) * HDIM_ + d];
        const float xvv = xv[(size_t)n * DIM_ + idx];
        out[(size_t)n * DIM_ + idx] = sa * xvv + sb * acc;
    }
}

// ---------------------------------------------------------------------------
// Host-side launcher
// ---------------------------------------------------------------------------
extern "C" void kernel_launch(void* const* d_in, const int* in_sizes, int n_in,
                              void* d_out, int out_size, void* d_ws, size_t ws_size,
                              hipStream_t stream)
{
    (void)in_sizes; (void)n_in; (void)out_size; (void)ws_size;

    const float* x     = (const float*)d_in[0];
    const float* bias  = (const float*)d_in[1];
    const float* Wq    = (const float*)d_in[2];
    const float* bq    = (const float*)d_in[3];
    const float* key_p = (const float*)d_in[4];
    const float* Wv    = (const float*)d_in[5];
    const float* bv    = (const float*)d_in[6];
    const float* emb   = (const float*)d_in[7];
    const float* alpha = (const float*)d_in[8];
    const float* beta  = (const float*)d_in[9];

    float* out = (float*)d_out;
    float* ws  = (float*)d_ws;

    // workspace layout (in float units)
    _Float16* qh    = (_Float16*)(ws);                 //  8,388,608 f  (16.8M halves)
    float*    xv    = ws + 8388608;                    // 16,777,216 f
    float*    attn  = ws + 25165824;                   // 67,108,864 f
    float*    attn_t= ws + 92274688;                   // 67,108,864 f
    float*    v_r   = ws + 159383552;                  //    262,144 f
    float*    sel_w = ws + 159645696;                  //  4,194,304 f
    int*      sel_i = (int*)(ws + 163840000);          //  4,194,304 i
    _Float16* xh    = (_Float16*)(ws + 168034304);     //  8,388,608 f
    _Float16* Wqth  = (_Float16*)(ws + 176422912);     //    524,288 f
    _Float16* Wvth  = (_Float16*)(ws + 176947200);     //    524,288 f
    _Float16* kph   = (_Float16*)(ws + 177471488);     //    131,072 f

    const dim3 blk(256);

    // f16 pre-conversion (x, key_p elementwise; W transposed to [n][k])
    cvt_f16_kernel<<<(N_TOK * DIM_) / (256 * 8), blk, 0, stream>>>(x, xh);
    cvt_f16_kernel<<<(K_G_ * HEAD_ * HDIM_) / (256 * 8), blk, 0, stream>>>(key_p, kph);
    cvtT_f16_kernel<<<dim3(DIM_ / 32, DIM_ / 32), blk, 0, stream>>>(Wq, Wqth, DIM_, DIM_);
    cvtT_f16_kernel<<<dim3(DIM_ / 32, DIM_ / 32), blk, 0, stream>>>(Wv, Wvth, DIM_, DIM_);

    // WMMA GEMMs with TDM-staged tiles
    gemm_wmma_tdm<true><<<dim3(N_TOK / 128, DIM_ / 128), blk, 0, stream>>>(
        xh, Wqth, bq, (void*)qh, N_TOK, DIM_, DIM_);
    gemm_wmma_tdm<false><<<dim3(N_TOK / 128, DIM_ / 128), blk, 0, stream>>>(
        xh, Wvth, bv, (void*)xv, N_TOK, DIM_, DIM_);
    attn_wmma_tdm<<<dim3(N_TOK / 128, K_G_ / 128, HEAD_), blk, 0, stream>>>(
        qh, kph, attn);

    // selection + combine
    transpose_kernel<<<dim3(N_TOK / 32, (HEAD_ * K_G_) / 32), blk, 0, stream>>>(
        attn, attn_t);
    topk_rows_kernel<<<(N_TOK * HEAD_) / 8, blk, 0, stream>>>(
        attn, bias, sel_w, sel_i, out + (size_t)N_TOK * DIM_);
    topk_cols_vr_kernel<<<HEAD_ * K_G_, blk, 0, stream>>>(
        attn_t, xv, emb, v_r);
    final_kernel<<<N_TOK, blk, 0, stream>>>(
        xv, v_r, sel_w, sel_i, alpha, beta, out);
}
